// GNN_57260503990332
// MI455X (gfx1250) — compile-verified, matrix operand order checked by reading, not run on hardware
//
#include <hip/hip_runtime.h>
#include <hip/hip_bf16.h>
#include <math.h>

typedef __attribute__((ext_vector_type(16))) _Float16 v16h;
typedef __attribute__((ext_vector_type(8)))  _Float16 v8h;
typedef __attribute__((ext_vector_type(8)))  float    v8f;

__device__ __forceinline__ float gelu_f(float x){
  return 0.5f * x * (1.0f + erff(x * 0.70710678118654752440f));
}

// A fragment (16x32 f16) from LDS, row stride 256 halves.
// Per ISA layout: lane L: m = L&15, kg = L>>4; slot i: k = kBase + i + 8*kg + (i>=8 ? 8 : 0)
__device__ __forceinline__ v16h load_a_frag(const _Float16* in, int rowBase, int kBase, int lane){
  int m  = rowBase + (lane & 15);
  int kg = lane >> 4;
  const _Float16* p = in + m * 256 + kBase + 8 * kg;
  v8h lo = *(const v8h*)(p);
  v8h hi = *(const v8h*)(p + 16);
  return __builtin_shufflevector(lo, hi, 0,1,2,3,4,5,6,7,8,9,10,11,12,13,14,15);
}

// 32 rows x 128 cols GEMM tile: D = A(32xK) * W(Kx128), accumulate f32.
// W pre-swizzled into B-fragment order: [(kt*8+nt)*32 + lane][16 halves] -> one 32B load/lane.
template<int K>
__device__ __forceinline__ void mma_tiles(const _Float16* in, const _Float16* w, int lane, v8f acc[2][8]){
  const v8f z = {0.f,0.f,0.f,0.f,0.f,0.f,0.f,0.f};
  #pragma unroll
  for (int mt=0; mt<2; ++mt)
    #pragma unroll
    for (int nt=0; nt<8; ++nt) acc[mt][nt] = z;
  #pragma unroll
  for (int kt=0; kt<K/32; ++kt){
    v16h a0 = load_a_frag(in, 0,  kt*32, lane);
    v16h a1 = load_a_frag(in, 16, kt*32, lane);
    const _Float16* wk = w + (long)kt * (8*32*16);
    #pragma unroll
    for (int nt=0; nt<8; ++nt){
      v16h b = *(const v16h*)(wk + (nt*32 + lane)*16);
      acc[0][nt] = __builtin_amdgcn_wmma_f32_16x16x32_f16(false, a0, false, b, (short)0, acc[0][nt], false, false);
      acc[1][nt] = __builtin_amdgcn_wmma_f32_16x16x32_f16(false, a1, false, b, (short)0, acc[1][nt], false, false);
    }
  }
}

// C layout: lane L: n = nt*16 + (L&15); vgpr r: m = mt*16 + (L>>4)*8 + r
__device__ __forceinline__ void store_act(_Float16* out, const float* __restrict__ bias,
                                          v8f acc[2][8], int lane, bool doGelu){
  int nl = lane & 15, kg = lane >> 4;
  #pragma unroll
  for (int mt=0; mt<2; ++mt)
    #pragma unroll
    for (int nt=0; nt<8; ++nt){
      int n = nt*16 + nl;
      float bv = bias[n];
      #pragma unroll
      for (int r=0; r<8; ++r){
        int m = mt*16 + kg*8 + r;
        float xv = acc[mt][nt][r] + bv;
        if (doGelu) xv = gelu_f(xv);
        out[m*256 + n] = (_Float16)xv;
      }
    }
}

// ---- weight pre-swizzle into B-fragment order (f16), Nc = 128, zero-padded K ----
__global__ void swizzle_kernel(const float* __restrict__ src, _Float16* __restrict__ dst,
                               int Kd, int Ks){
  int total = Kd * 128;
  for (int idx = blockIdx.x*blockDim.x + threadIdx.x; idx < total; idx += gridDim.x*blockDim.x){
    int i    = idx & 15;
    int lane = (idx >> 4) & 31;
    int fn   = idx >> 9;        // kt*8 + nt
    int nt = fn & 7;
    int kt = fn >> 3;
    int kg = lane >> 4, nl = lane & 15;
    int k = kt*32 + i + 8*kg + ((i>=8)?8:0);
    int n = nt*16 + nl;
    dst[idx] = (k < Ks) ? (_Float16)src[k*128 + n] : (_Float16)0.f;
  }
}

// ---- encoder: WMMA layer0 (K zero-padded to 32) + 3 WMMA layers ----
template<int DIN, bool EDGE>
__global__ __launch_bounds__(128) void encode_kernel(
    const float* __restrict__ xin, long long R,
    const _Float16* __restrict__ w0, const float* __restrict__ b0,
    const _Float16* __restrict__ w1, const float* __restrict__ b1,
    const _Float16* __restrict__ w2, const float* __restrict__ b2,
    const _Float16* __restrict__ w3, const float* __restrict__ b3,
    float* __restrict__ vout, _Float16* __restrict__ ehout, float* __restrict__ stats)
{
  __shared__ __align__(16) _Float16 act[4][32][256];
  int wave = threadIdx.x >> 5, lane = threadIdx.x & 31;
  long long base = ((long long)blockIdx.x * 4 + wave) * 32;
  _Float16* A = &act[wave][0][0];

  { // input row -> f16, zero-padded to 32 cols
    long long row = base + lane;
    bool ok = row < R;
    #pragma unroll
    for (int c=0;c<32;c++){
      float xv = (ok && c < DIN) ? xin[row*DIN + c] : 0.f;
      A[lane*256 + c] = (_Float16)xv;
    }
  }
  __syncthreads();
  v8f acc[2][8];
  mma_tiles<32> (A + 0,   w0, lane, acc); store_act(A + 128, b0, acc, lane, true);
  __syncthreads();
  mma_tiles<128>(A + 128, w1, lane, acc); store_act(A + 0,   b1, acc, lane, true);
  __syncthreads();
  mma_tiles<128>(A + 0,   w2, lane, acc); store_act(A + 128, b2, acc, lane, true);
  __syncthreads();
  mma_tiles<128>(A + 128, w3, lane, acc);

  int nl = lane & 15, kg = lane >> 4;
  #pragma unroll
  for (int mt=0; mt<2; ++mt)
    #pragma unroll
    for (int nt=0; nt<8; ++nt){
      int n = nt*16 + nl;
      float bv = b3[n];
      float s = 0.f, s2 = 0.f;
      #pragma unroll
      for (int r=0;r<8;++r){
        int m = mt*16 + kg*8 + r;
        long long row = base + m;
        float xv = acc[mt][nt][r] + bv;
        if (row < R){
          if (EDGE){ ehout[row*128 + n] = (_Float16)xv; s += xv; s2 += xv*xv; }
          else     { vout[row*128 + n] = xv; }
        }
      }
      if (EDGE){
        s  += __shfl_xor(s, 16);
        s2 += __shfl_xor(s2, 16);
        if (kg == 0){ atomicAdd(&stats[n], s); atomicAdd(&stats[128+n], s2); }
      }
    }
}

// ---- per-round node BN ----
__global__ __launch_bounds__(128) void col_stats_kernel(const float* __restrict__ v, long long N,
                                                        float* __restrict__ stats){
  int c = threadIdx.x;
  long long r0 = (long long)blockIdx.x * 256;
  float s=0.f, s2=0.f;
  for (int i=0;i<256;i++){
    long long r = r0 + i;
    if (r < N){ float x = v[r*128 + c]; s += x; s2 += x*x; }
  }
  atomicAdd(&stats[c], s); atomicAdd(&stats[128+c], s2);
}

__global__ void finalize_bn_kernel(const float* __restrict__ stats, float cnt,
                                   const float* __restrict__ gamma, const float* __restrict__ beta,
                                   float* __restrict__ ss, _Float16* __restrict__ ssh){
  int c = threadIdx.x;
  float mean = stats[c] / cnt;
  float var  = stats[128+c] / cnt - mean*mean;
  float sc = gamma[c] * rsqrtf(var + 1e-5f);
  float sh = beta[c] - mean*sc;
  ss[c] = sc; ss[128+c] = sh;
  if (ssh){ ssh[c] = (_Float16)sc; ssh[128+c] = (_Float16)sh; }
}

__global__ __launch_bounds__(256) void bn_apply_kernel(const float* __restrict__ v, const float* __restrict__ ss,
    _Float16* __restrict__ vbn, float* __restrict__ vnext, long long total){
  long long i = (long long)blockIdx.x*blockDim.x + threadIdx.x;
  long long stride = (long long)gridDim.x * blockDim.x;
  for (; i < total; i += stride){
    int c = (int)(i & 127);
    float x = v[i]*ss[c] + ss[128+c];
    vbn[i] = (_Float16)x;   // f16 copy for edge gathers (halves gather BW)
    vnext[i] = x;           // f32 residual base, messages atomically added here
  }
}

// ---- message MLP: packed-f16 gather -> 256->128->128->128->128 WMMA -> atomic scatter ----
__global__ __launch_bounds__(128) void msg_kernel(
    const _Float16* __restrict__ vbn, const _Float16* __restrict__ eh,
    const _Float16* __restrict__ essh, const int* __restrict__ ei, long long E,
    const _Float16* __restrict__ w0, const float* __restrict__ b0,
    const _Float16* __restrict__ w1, const float* __restrict__ b1,
    const _Float16* __restrict__ w2, const float* __restrict__ b2,
    const _Float16* __restrict__ w3, const float* __restrict__ b3,
    float* __restrict__ vnext)
{
  __shared__ __align__(16) _Float16 act[4][32][256];
  int wave = threadIdx.x >> 5, lane = threadIdx.x & 31;
  long long base = ((long long)blockIdx.x * 4 + wave) * 32;
  _Float16* A = &act[wave][0][0];

  { // gather: lane owns one edge-row; packed f16 math (v_pk_add/fma_f16), b128 loads
    long long e = base + lane;
    v8h* A8 = (v8h*)(A + lane*256);
    if (e < E){
      int s = ei[e];
      int d = ei[E + e];
      const v8h* ps8 = (const v8h*)(vbn + (long long)s * 128);
      const v8h* pd8 = (const v8h*)(vbn + (long long)d * 128);
      const v8h* pe8 = (const v8h*)(eh + e * 128);
      const v8h* sc8 = (const v8h*)essh;      // [0..15]=scale, [16..31]=shift
      #pragma unroll
      for (int c=0;c<16;c++) A8[c] = pd8[c] + ps8[c];
      #pragma unroll
      for (int c=0;c<16;c++) A8[16+c] = pe8[c]*sc8[c] + sc8[16+c];
    } else {
      const v8h z = {};
      #pragma unroll
      for (int c=0;c<32;c++) A8[c] = z;
    }
  }
  __syncthreads();
  v8f acc[2][8];
  mma_tiles<256>(A + 0,   w0, lane, acc); store_act(A + 0,   b0, acc, lane, true);
  __syncthreads();
  mma_tiles<128>(A + 0,   w1, lane, acc); store_act(A + 128, b1, acc, lane, true);
  __syncthreads();
  mma_tiles<128>(A + 128, w2, lane, acc); store_act(A + 0,   b2, acc, lane, true);
  __syncthreads();
  mma_tiles<128>(A + 0,   w3, lane, acc);

  // segment_sum: f32 atomic scatter into residual buffer
  int nl = lane & 15, kg = lane >> 4;
  #pragma unroll
  for (int mt=0; mt<2; ++mt)
    #pragma unroll
    for (int nt=0; nt<8; ++nt){
      int n = nt*16 + nl;
      float bv = b3[n];
      #pragma unroll
      for (int r=0;r<8;++r){
        int m = mt*16 + kg*8 + r;
        long long e = base + m;
        if (e < E){
          int d = ei[E + e];
          atomicAdd(vnext + (long long)d*128 + n, acc[mt][nt][r] + bv);
        }
      }
    }
}

// ---- decoder: 3x WMMA + scalar 128->3 ----
__global__ __launch_bounds__(128) void dec_kernel(
    const float* __restrict__ v, long long N,
    const _Float16* __restrict__ w0, const float* __restrict__ b0,
    const _Float16* __restrict__ w1, const float* __restrict__ b1,
    const _Float16* __restrict__ w2, const float* __restrict__ b2,
    const float* __restrict__ W3, const float* __restrict__ b3,
    float* __restrict__ out)
{
  __shared__ __align__(16) _Float16 act[4][32][256];
  int wave = threadIdx.x >> 5, lane = threadIdx.x & 31;
  long long base = ((long long)blockIdx.x * 4 + wave) * 32;
  _Float16* A = &act[wave][0][0];
  long long row = base + lane;
  if (row < N){ for (int c=0;c<128;c++) A[lane*256 + c] = (_Float16)v[row*128 + c]; }
  else        { for (int c=0;c<128;c++) A[lane*256 + c] = (_Float16)0.f; }
  __syncthreads();
  v8f acc[2][8];
  mma_tiles<128>(A + 0,   w0, lane, acc); store_act(A + 128, b0, acc, lane, true);
  __syncthreads();
  mma_tiles<128>(A + 128, w1, lane, acc); store_act(A + 0,   b1, acc, lane, true);
  __syncthreads();
  mma_tiles<128>(A + 0,   w2, lane, acc); store_act(A + 128, b2, acc, lane, true);
  __syncthreads();
  if (row < N){
    float o0 = b3[0], o1 = b3[1], o2 = b3[2];
    for (int c=0;c<128;c++){
      float h = (float)A[lane*256 + 128 + c];
      o0 += h * W3[c*3+0]; o1 += h * W3[c*3+1]; o2 += h * W3[c*3+2];
    }
    out[row*3+0]=o0; out[row*3+1]=o1; out[row*3+2]=o2;
  }
}

extern "C" void kernel_launch(void* const* d_in, const int* in_sizes, int n_in,
                              void* d_out, int out_size, void* d_ws, size_t ws_size,
                              hipStream_t stream)
{
  (void)in_sizes; (void)n_in; (void)out_size; (void)ws_size;
  const long long N = 50000, E = 600000;
  const float* x  = (const float*)d_in[0];
  const float* ea = (const float*)d_in[1];
  const int*   ei = (const int*)d_in[2];
  auto F = [&](int i){ return (const float*)d_in[i]; };

  // workspace layout (bytes)
  char* ws = (char*)d_ws;
  float*    v0   = (float*)(ws + 0LL);            // N*128 f32
  float*    v1   = (float*)(ws + 25600000LL);     // N*128 f32
  _Float16* vbn  = (_Float16*)(ws + 51200000LL);  // N*128 f16
  _Float16* eh   = (_Float16*)(ws + 64000000LL);  // E*128 f16
  _Float16* wsw  = (_Float16*)(ws + 217600000LL); // swizzled weights, 401408 halves
  float*    nstats = (float*)(ws + 218500000LL);  // 256
  float*    estats = nstats + 256;                // 256
  float*    nss    = estats + 256;                // 256 (scale|shift)
  float*    essf   = nss + 256;                   // 256 f32 (unused by msg, kept)
  _Float16* essh   = (_Float16*)(essf + 256);     // 256 f16 (scale|shift)

  // 1) pre-swizzle all WMMA weight matrices into B-fragment order (f16, L2-resident)
  //    layer-0 matrices are zero-padded from K=DIN to K=32
  struct SwRec { int src; long off; int Kd; int Ks; };
  const SwRec tbl[24] = {
    {3,      0, 32,   6},{5,   4096,128,128},{7,  20480,128,128},{9,  36864,128,128}, // node_enc
    {11, 53248, 32,   4},{13, 57344,128,128},{15, 73728,128,128},{17, 90112,128,128}, // edge_enc
    {19,106496,256, 256},{21,139264,128,128},{23,155648,128,128},{25,172032,128,128}, // mp0
    {27,188416,256, 256},{29,221184,128,128},{31,237568,128,128},{33,253952,128,128}, // mp1
    {35,270336,256, 256},{37,303104,128,128},{39,319488,128,128},{41,335872,128,128}, // mp2
    {43,352256,128, 128},{45,368640,128,128},{47,385024,128,128},{-1,0,0,0}
  };
  for (int t=0;t<23;t++){
    int total = tbl[t].Kd * 128;
    swizzle_kernel<<<(total+255)/256, 256, 0, stream>>>(F(tbl[t].src), wsw + tbl[t].off,
                                                        tbl[t].Kd, tbl[t].Ks);
  }

  int eb = (int)((E + 127) / 128);
  int nb = (int)((N + 127) / 128);

  // 2) edge encoder + BN stats, then edge BN scale/shift (f32 + f16 copies)
  hipMemsetAsync(estats, 0, 256*sizeof(float), stream);
  encode_kernel<4,true><<<eb,128,0,stream>>>(ea, E,
      wsw+53248, F(12), wsw+57344, F(14), wsw+73728, F(16), wsw+90112, F(18),
      nullptr, eh, estats);
  finalize_bn_kernel<<<1,128,0,stream>>>(estats, (float)E, F(51), F(52), essf, essh);

  // 3) node encoder
  encode_kernel<6,false><<<nb,128,0,stream>>>(x, N,
      wsw+0, F(4), wsw+4096, F(6), wsw+20480, F(8), wsw+36864, F(10),
      v0, nullptr, nullptr);

  // 4) three message-passing rounds
  float* vc = v0; float* vn = v1;
  for (int r=0;r<3;r++){
    hipMemsetAsync(nstats, 0, 256*sizeof(float), stream);
    col_stats_kernel<<<(int)((N+255)/256),128,0,stream>>>(vc, N, nstats);
    finalize_bn_kernel<<<1,128,0,stream>>>(nstats, (float)N, F(51), F(52), nss, nullptr);
    bn_apply_kernel<<<2048,256,0,stream>>>(vc, nss, vbn, vn, N*128);
    const _Float16* w0 = wsw + 106496 + (long)r*81920;
    msg_kernel<<<eb,128,0,stream>>>(vbn, eh, essh, ei, E,
        w0,        F(20+8*r),
        w0+32768,  F(22+8*r),
        w0+49152,  F(24+8*r),
        w0+65536,  F(26+8*r),
        vn);
    float* t = vc; vc = vn; vn = t;
  }

  // 5) decoder -> d_out [N,3] f32
  dec_kernel<<<nb,128,0,stream>>>(vc, N,
      wsw+352256, F(44), wsw+368640, F(46), wsw+385024, F(48),
      F(49), F(50), (float*)d_out);
}
